// MBartGQAttention_12446815223984
// MI455X (gfx1250) — matrix operand order repaired and checked by reading, not compiled, not run on hardware
//
#include <hip/hip_runtime.h>
#include <hip/hip_bf16.h>

// ---------------------------------------------------------------------------
// MBart GQA prefill attention for MI455X (gfx1250, wave32, WMMA).
// All matmuls (Q/K/V proj, QK^T, PV, O proj) run on v_wmma_f32_16x16x32_bf16.
// V tiles are staged to LDS with global_load_async_to_lds_b128 (ASYNCcnt),
// upcoming tiles are prefetched branchlessly with global_prefetch_b8.
// ---------------------------------------------------------------------------

#define BSZ      2
#define SEQ      2048
#define EMB      1024
#define NHEADS   16
#define NKVH     4
#define HDIM     64
#define KVDIM    (NKVH * HDIM)      // 256
#define MROWS    (BSZ * SEQ)        // 4096
#define SCALING  0.125f             // 64^-0.5

typedef __attribute__((ext_vector_type(16))) __bf16 v16bf;
typedef __attribute__((ext_vector_type(8)))  __bf16 v8bf;
typedef __attribute__((ext_vector_type(8)))  float  v8f;

// ---- helpers ---------------------------------------------------------------

__device__ __forceinline__ __bf16 f2bf(float f) {
    unsigned int u = __builtin_bit_cast(unsigned int, f);
    unsigned int r = (u + 0x7fffu + ((u >> 16) & 1u)) >> 16;
    return __builtin_bit_cast(__bf16, (unsigned short)r);
}

__device__ __forceinline__ v16bf cat8(v8bf a, v8bf b) {
    return __builtin_shufflevector(a, b, 0,1,2,3,4,5,6,7,8,9,10,11,12,13,14,15);
}

__device__ __forceinline__ v8f wmma_bf16(v16bf a, v16bf b, v8f c) {
    // D = A(16x32 bf16) * B(32x16 bf16) + C(16x16 f32)
    return __builtin_amdgcn_wmma_f32_16x16x32_bf16(
        false, a, false, b, (short)0, c, false, false);
}

// A fragment: 16x32 tile of a row-major [M x lda] bf16 matrix at (row0, k0).
// Layout used here (contiguous k, same as B): lane m = L%16, hi = L/16; slots 0..15 = k0+16*hi+(0..15)
// contiguous (loaded as two 8-element runs).
__device__ __forceinline__ v16bf load_a_frag(const __bf16* A, int lda,
                                             int row0, int k0) {
    int lane = threadIdx.x & 31;
    int m = lane & 15, hi = lane >> 4;
    const __bf16* p = A + (size_t)(row0 + m) * lda + k0 + hi * 16;
    v8bf lo = *(const v8bf*)p;
    v8bf hv = *(const v8bf*)(p + 8);
    return cat8(lo, hv);
}

// B fragment from a pre-transposed Bt[N x K] (row-major) at (n0, k0).
// Layout used here (contiguous k, same as A): lane n = L%16, hi = L/16; slots = k0+16*hi+(0..15) contiguous.
__device__ __forceinline__ v16bf load_bt_frag(const __bf16* Bt, int ldb,
                                              int n0, int k0) {
    int lane = threadIdx.x & 31;
    int n = lane & 15, hi = lane >> 4;
    return *(const v16bf*)(Bt + (size_t)(n0 + n) * ldb + k0 + hi * 16);
}

// ---- conversion kernels ----------------------------------------------------

__global__ __launch_bounds__(256) void cvt_bf16_kernel(const float* __restrict__ in,
                                                       __bf16* __restrict__ out, int n) {
    int i = blockIdx.x * blockDim.x + threadIdx.x;
    if (i < n) out[i] = f2bf(in[i]);
}

// in: f32 [K x N] row-major  ->  out: bf16 [N x K] row-major (transpose)
__global__ __launch_bounds__(256) void cvt_tr_kernel(const float* __restrict__ in,
                                                     __bf16* __restrict__ out,
                                                     int K, int N) {
    int i = blockIdx.x * blockDim.x + threadIdx.x;
    if (i < K * N) {
        int k = i / N, n = i % N;
        out[(size_t)n * K + k] = f2bf(in[i]);
    }
}

// ---- generic WMMA GEMM: C = (A @ B + bias) * scale -------------------------
// A: bf16 [M x K] row-major.  Bt: bf16 [N x K] (B transposed).
// outBf (optional): bf16 [M x N] row-major.
// f32mode: 0 = none, 1 = f32 [M x N] row-major (final out),
//          2 = past_key_value layout [b][kvh][t][d] (outF32 pre-offset by kv slot)
__global__ __launch_bounds__(256) void wmma_gemm_kernel(
    const __bf16* __restrict__ A, const __bf16* __restrict__ Bt,
    const float* __restrict__ bias, float scale,
    __bf16* __restrict__ outBf, float* __restrict__ outF32, int f32mode,
    int M, int N, int K) {

    int w  = threadIdx.x >> 5;                 // wave id 0..7
    int mt = blockIdx.y * 2 + (w >> 2);        // block tile: 32 rows x 64 cols
    int nt = blockIdx.x * 4 + (w & 3);
    int row0 = mt * 16, n0 = nt * 16;
    int lane = threadIdx.x & 31;
    int lm = lane & 15;

    const __bf16* arow = A  + (size_t)(row0 + lm) * K;
    const __bf16* brow = Bt + (size_t)(n0   + lm) * K;

    v8f acc = {};
#pragma unroll 8
    for (int k0 = 0; k0 < K; k0 += 32) {
        // Branchless amortized prefetch: (k0 & 255)==0 constant-folds per
        // unrolled copy, so this emits once per 8 iterations with no branch.
        // Speculative DEV-scope prefetch past the matrix end is silently
        // dropped by hardware (and stays inside the workspace regardless).
        if ((k0 & 255) == 0) {
            __builtin_prefetch(arow + k0 + 256, 0, 1);
            __builtin_prefetch(arow + k0 + 320, 0, 1);
            __builtin_prefetch(arow + k0 + 384, 0, 1);
            __builtin_prefetch(arow + k0 + 448, 0, 1);
            __builtin_prefetch(brow + k0 + 256, 0, 1);
            __builtin_prefetch(brow + k0 + 320, 0, 1);
            __builtin_prefetch(brow + k0 + 384, 0, 1);
            __builtin_prefetch(brow + k0 + 448, 0, 1);
        }
        v16bf a = load_a_frag(A, K, row0, k0);
        v16bf b = load_bt_frag(Bt, K, n0, k0);
        acc = wmma_bf16(a, b, acc);
    }

    int n = lm, part = lane >> 4;
    int col = n0 + n;
    float bv = bias[col];

#pragma unroll
    for (int r = 0; r < 8; ++r) {
        float v   = (acc[r] + bv) * scale;
        int   row = row0 + r + 8 * part;       // C layout: M = r + 8*(lane/16)
        if (outBf) outBf[(size_t)row * N + col] = f2bf(v);
        if (f32mode == 1) {
            outF32[(size_t)row * N + col] = v;
        } else if (f32mode == 2) {
            int b_ = row / SEQ, t = row % SEQ;
            int kvh = col >> 6, d = col & 63;
            outF32[(((size_t)b_ * NKVH + kvh) * SEQ + t) * HDIM + d] = v;
        }
    }
}

// ---- flash attention: one wave per (b, h, 16-row q tile) -------------------
// Q: bf16 [B*T x EMB] (already scaled, head h at col h*64).
// Kb/Vb: bf16 [B*T x KVDIM] (kv head at col kvh*64).
// attnOut: bf16 [B*T x EMB], laid out as the A matrix of the O projection.
__global__ __launch_bounds__(32) void attn_kernel(
    const __bf16* __restrict__ Q, const __bf16* __restrict__ Kb,
    const __bf16* __restrict__ Vb, __bf16* __restrict__ attnOut) {

    __shared__ __align__(32) __bf16 lds_p[16 * 32];   // P tile  [m][s_local]
    __shared__ __align__(32) __bf16 lds_v[32 * 64];   // V tile  [s_local][d]

    int tile = blockIdx.x;                 // q tile (0..127)
    int h    = blockIdx.y;                 // head
    int b    = blockIdx.z;                 // batch
    int kvh  = h >> 2;                     // GQA group
    int t0   = tile * 16;

    int lane = threadIdx.x & 31;
    int n = lane & 15, part = lane >> 4;

    const __bf16* Qb = Q  + (size_t)b * SEQ * EMB   + h   * HDIM;
    const __bf16* Kr = Kb + (size_t)b * SEQ * KVDIM + kvh * HDIM;
    const __bf16* Vr = Vb + (size_t)b * SEQ * KVDIM + kvh * HDIM;

    // Q A-fragments for d=0..31 and d=32..63, resident for the whole pass.
    v16bf qa0 = load_a_frag(Qb, EMB, t0, 0);
    v16bf qa1 = load_a_frag(Qb, EMB, t0, 32);

    float mrow[8], lrow[8];
    v8f o[4] = {};
#pragma unroll
    for (int r = 0; r < 8; ++r) { mrow[r] = -1e30f; lrow[r] = 0.0f; }

    int s_end = t0 + 16;                   // causal: s <= t
    for (int s0 = 0; s0 < s_end; s0 += 32) {
        // ---- async DMA: V chunk [32 x 64] -> LDS (natural layout) ----------
        // Each lane copies its own 128B row in 8 b128 transfers (ASYNCcnt).
        {
            unsigned ldsOff = (unsigned)(size_t)(&lds_v[0]) + (unsigned)lane * 128u;
            unsigned long long ga =
                (unsigned long long)(size_t)(Vr + (size_t)(s0 + lane) * KVDIM);
#pragma unroll
            for (int q = 0; q < 8; ++q) {
                unsigned lo = ldsOff + (unsigned)(q * 16);
                unsigned long long g = ga + (unsigned long long)(q * 16);
                asm volatile("global_load_async_to_lds_b128 %0, %1, off"
                             :: "v"(lo), "v"(g) : "memory");
            }
        }

        // ---- prefetch next chunk's K rows while this chunk computes --------
        if (s0 + 32 < s_end) {
            __builtin_prefetch(Kr + (size_t)(s0 + 32 + n) * KVDIM, 0, 1);
            __builtin_prefetch(Kr + (size_t)(s0 + 48 + n) * KVDIM, 0, 1);
        }

        // ---- scores S[16t x 32s] = Q @ K^T (two 16x16 WMMA tiles) ----------
        v8f S0 = {}, S1 = {};
        {
            const __bf16* kp = Kr + (size_t)(s0 + n) * KVDIM;
            v16bf kb0 = *(const v16bf*)(kp + part * 16);        // d 0..31
            v16bf kb1 = *(const v16bf*)(kp + 32 + part * 16);   // d 32..63
            S0 = wmma_bf16(qa0, kb0, S0);
            S0 = wmma_bf16(qa1, kb1, S0);
            const __bf16* kp2 = Kr + (size_t)(s0 + 16 + n) * KVDIM;
            v16bf kb2 = *(const v16bf*)(kp2 + part * 16);
            v16bf kb3 = *(const v16bf*)(kp2 + 32 + part * 16);
            S1 = wmma_bf16(qa0, kb2, S1);
            S1 = wmma_bf16(qa1, kb3, S1);
        }

        // ---- causal mask + online softmax (rows spread over 16 lanes) ------
        float cf[8];
#pragma unroll
        for (int r = 0; r < 8; ++r) {
            int trow = t0 + r + 8 * part;
            if (s0 + n      > trow) S0[r] = -1e30f;
            if (s0 + 16 + n > trow) S1[r] = -1e30f;

            float cmax = fmaxf(S0[r], S1[r]);
            cmax = fmaxf(cmax, __shfl_xor(cmax, 1, 16));
            cmax = fmaxf(cmax, __shfl_xor(cmax, 2, 16));
            cmax = fmaxf(cmax, __shfl_xor(cmax, 4, 16));
            cmax = fmaxf(cmax, __shfl_xor(cmax, 8, 16));

            float mnew = fmaxf(mrow[r], cmax);
            cf[r]   = __expf(mrow[r] - mnew);
            mrow[r] = mnew;

            float p0 = __expf(S0[r] - mnew);
            float p1 = __expf(S1[r] - mnew);
            S0[r] = p0; S1[r] = p1;

            float rs = p0 + p1;
            rs += __shfl_xor(rs, 1, 16);
            rs += __shfl_xor(rs, 2, 16);
            rs += __shfl_xor(rs, 4, 16);
            rs += __shfl_xor(rs, 8, 16);
            lrow[r] = lrow[r] * cf[r] + rs;
        }

        // ---- rescale running O accumulators --------------------------------
#pragma unroll
        for (int dt = 0; dt < 4; ++dt)
#pragma unroll
            for (int r = 0; r < 8; ++r) o[dt][r] *= cf[r];

        // ---- stage P (C layout -> A layout) through LDS --------------------
#pragma unroll
        for (int r = 0; r < 8; ++r) {
            int m = r + 8 * part;
            lds_p[m * 32 + n]      = f2bf(S0[r]);
            lds_p[m * 32 + 16 + n] = f2bf(S1[r]);
        }

        // wait for the async V DMA, then make all LDS writes visible
        asm volatile("s_wait_asynccnt 0" ::: "memory");
        __syncthreads();

        // ---- O += P(16x32) @ V(32x64), four 16x16 d-tiles ------------------
        v16bf pa;
        {
            const __bf16* pp = lds_p + n * 32 + part * 16;  // m'=lane%16, hi=lane/16
            v8bf lo = *(const v8bf*)pp;
            v8bf hv = *(const v8bf*)(pp + 8);
            pa = cat8(lo, hv);
        }
#pragma unroll
        for (int dt = 0; dt < 4; ++dt) {
            v16bf vb;
#pragma unroll
            for (int i = 0; i < 16; ++i)           // B layout: k = 16*hi + i
                vb[i] = lds_v[(part * 16 + i) * 64 + dt * 16 + n];
            o[dt] = wmma_bf16(pa, vb, o[dt]);
        }
        __syncthreads();
    }

    // ---- normalize and write attn tile as bf16 [b*T+t][h*64+d] -------------
#pragma unroll
    for (int r = 0; r < 8; ++r) {
        float inv  = 1.0f / lrow[r];
        int   trow = t0 + r + 8 * part;
        __bf16* dst = attnOut + ((size_t)b * SEQ + trow) * EMB + h * HDIM;
#pragma unroll
        for (int dt = 0; dt < 4; ++dt)
            dst[dt * 16 + n] = f2bf(o[dt][r] * inv);
    }
}

// ---------------------------------------------------------------------------

extern "C" void kernel_launch(void* const* d_in, const int* in_sizes, int n_in,
                              void* d_out, int out_size, void* d_ws, size_t ws_size,
                              hipStream_t stream) {
    const float* X  = (const float*)d_in[0];   // [2,2048,1024]
    // d_in[1] is the additive causal mask; we apply causality analytically.
    const float* Wq = (const float*)d_in[2];
    const float* bq = (const float*)d_in[3];
    const float* Wk = (const float*)d_in[4];
    const float* bk = (const float*)d_in[5];
    const float* Wv = (const float*)d_in[6];
    const float* bv = (const float*)d_in[7];
    const float* Wo = (const float*)d_in[8];
    const float* bo = (const float*)d_in[9];

    float* out = (float*)d_out;                               // [2,2048,1024]
    float* pkv = out + (size_t)BSZ * SEQ * EMB;               // [2,2,4,2048,64]
    const size_t kvPlane = (size_t)BSZ * NKVH * SEQ * HDIM;   // one of K/V

    // workspace carve-up (bf16 buffers, 256B aligned)
    char* w = (char*)d_ws;
    size_t off = 0;
    auto take = [&](size_t elems) {
        void* p = w + off;
        off = (off + elems * sizeof(__bf16) + 255) & ~(size_t)255;
        return p;
    };
    __bf16* Xbf  = (__bf16*)take((size_t)MROWS * EMB);
    __bf16* WqT  = (__bf16*)take((size_t)EMB * EMB);
    __bf16* WkT  = (__bf16*)take((size_t)EMB * KVDIM);
    __bf16* WvT  = (__bf16*)take((size_t)EMB * KVDIM);
    __bf16* WoT  = (__bf16*)take((size_t)EMB * EMB);
    __bf16* Qbf  = (__bf16*)take((size_t)MROWS * EMB);
    __bf16* Kbf  = (__bf16*)take((size_t)MROWS * KVDIM);
    __bf16* Vbf  = (__bf16*)take((size_t)MROWS * KVDIM);
    __bf16* Abf  = (__bf16*)take((size_t)MROWS * EMB);
    (void)ws_size; (void)in_sizes; (void)n_in; (void)out_size;

    // 1) activations -> bf16 ; weights -> transposed bf16
    {
        int n = MROWS * EMB;
        cvt_bf16_kernel<<<(n + 255) / 256, 256, 0, stream>>>(X, Xbf, n);
        int nq = EMB * EMB;
        cvt_tr_kernel<<<(nq + 255) / 256, 256, 0, stream>>>(Wq, WqT, EMB, EMB);
        int nk = EMB * KVDIM;
        cvt_tr_kernel<<<(nk + 255) / 256, 256, 0, stream>>>(Wk, WkT, EMB, KVDIM);
        cvt_tr_kernel<<<(nk + 255) / 256, 256, 0, stream>>>(Wv, WvT, EMB, KVDIM);
        cvt_tr_kernel<<<(nq + 255) / 256, 256, 0, stream>>>(Wo, WoT, EMB, EMB);
    }

    // 2) projections (WMMA GEMMs). Q fuses bias + 1/sqrt(d) scaling.
    {
        dim3 blk(256);
        dim3 gq(EMB / 64, MROWS / 32);     // N=1024
        wmma_gemm_kernel<<<gq, blk, 0, stream>>>(Xbf, WqT, bq, SCALING,
                                                 Qbf, nullptr, 0, MROWS, EMB, EMB);
        dim3 gkv(KVDIM / 64, MROWS / 32);  // N=256
        wmma_gemm_kernel<<<gkv, blk, 0, stream>>>(Xbf, WkT, bk, 1.0f,
                                                  Kbf, pkv, 2, MROWS, KVDIM, EMB);
        wmma_gemm_kernel<<<gkv, blk, 0, stream>>>(Xbf, WvT, bv, 1.0f,
                                                  Vbf, pkv + kvPlane, 2, MROWS, KVDIM, EMB);
    }

    // 3) causal flash attention
    {
        dim3 grid(SEQ / 16, NHEADS, BSZ);
        attn_kernel<<<grid, 32, 0, stream>>>(Qbf, Kbf, Vbf, Abf);
    }

    // 4) output projection -> fp32 d_out
    {
        dim3 blk(256);
        dim3 go(EMB / 64, MROWS / 32);
        wmma_gemm_kernel<<<go, blk, 0, stream>>>(Abf, WoT, bo, 1.0f,
                                                 nullptr, out, 1, MROWS, EMB, EMB);
    }
}
